// TBlock_33217277067817
// MI455X (gfx1250) — matrix operand order repaired and checked by reading, not compiled, not run on hardware
//
#include <hip/hip_runtime.h>

// Fused HaloNet block-local attention for gfx1250 (MI455X).
// One workgroup per (batch, nh, nw) window; 4 waves (wave32), one head per wave.
// Phase 0: cooperatively stage the 14x14 halo window of x into LDS as packed
//          bf16 channel-pairs (B-operand layout for v_wmma_f32_16x16x32_bf16).
// Phase A: per-head Q/K/V projections (W is fp32 in global, converted to bf16
//          A-tiles via v_cvt_pk_bf16_f32), rel bias folded into K, d^-0.5*log2e
//          folded into Q; TWO pixel-tiles processed concurrently per loop so the
//          WMMA accumulate chains interleave (1 wave/SIMD needs the ILP).
// Phase B: sim = Q@K^T (one wmma per 16x16 tile since d==32), masked base-2
//          softmax in f32 registers, P stored bf16, out^T = V^T @ P^T in f32,
//          written straight to the (b, c, h, w) output layout.

#define WIN   14
#define PIX   196        // WIN*WIN
#define PPAD  210        // padded pixel stride (u32 rows) for Xp / Kp
#define VSTR  226        // Vt row stride (u16), odd/2 -> conflict-free u32 rows
#define ASTR  226        // At row stride (u16)
#define QSTR  34         // Qb row stride (u16)
#define JPAD  224        // keys padded to 7*32 for the output GEMM
#define CH    128
#define DIM   32

typedef __attribute__((ext_vector_type(16))) __bf16        v16bf;
typedef __attribute__((ext_vector_type(8)))  float         v8f;
typedef __attribute__((ext_vector_type(8)))  unsigned int  v8u;

#define V8F_ZERO ((v8f){0.f,0.f,0.f,0.f,0.f,0.f,0.f,0.f})

// ---- float -> bf16: hardware packed convert (1 VALU op per pair) ----
static __device__ __forceinline__ unsigned packbf(float lo, float hi){
  unsigned r;
  asm("v_cvt_pk_bf16_f32 %0, %1, %2" : "=v"(r) : "v"(lo), "v"(hi));
  return r;
}
// convert a pair and store to two (non-adjacent) u16 slots
static __device__ __forceinline__ void store2_bf16(unsigned short* p0,
                                                   unsigned short* p1,
                                                   float a, float b){
  const unsigned pk = packbf(a, b);
  *p0 = (unsigned short)pk;           // ds_store_b16
  *p1 = (unsigned short)(pk >> 16);   // ds_store_b16_d16_hi
}

// ---- raw base-2 exp (v_exp_f32); log2(e) is folded into the Q scale ----
#if __has_builtin(__builtin_amdgcn_exp2f)
#define EXP2F(x) __builtin_amdgcn_exp2f(x)
#else
#define EXP2F(x) exp2f(x)
#endif

static __device__ __forceinline__ v8f wmma_bf16(v8u a, v8u b, v8f c){
  return __builtin_amdgcn_wmma_f32_16x16x32_bf16(
      false, __builtin_bit_cast(v16bf, a),
      false, __builtin_bit_cast(v16bf, b),
      (short)0, c, false, false);
}

// A-tile (16x32 bf16) from fp32 weights in global memory.
// Layout: lane<16 -> M=lane, K in {k0..k0+7} u {k0+16..k0+23} as packed pairs;
//         lane>=16 -> same M, K shifted by +8.
static __device__ __forceinline__ v8u loadA_W(const float* __restrict__ W,
                                              int row0, int k0, int lane){
  const int row = row0 + (lane & 15);
  const int kb  = k0 + ((lane >> 4) << 3);
  const float2* wr = (const float2*)(W + row * CH + kb);
  v8u a;
#pragma unroll
  for (int t = 0; t < 4; ++t){ float2 f = wr[t];     a[t]     = packbf(f.x, f.y); }
#pragma unroll
  for (int t = 0; t < 4; ++t){ float2 f = wr[8 + t]; a[4 + t] = packbf(f.x, f.y); }
  return a;
}

// B-tile (32x16 bf16) from LDS packed channel-pairs. p = this lane's column pixel.
static __device__ __forceinline__ v8u loadB_Xp(const unsigned* __restrict__ Xp,
                                               int c0, int p, int lane){
  const int cpb = (c0 >> 1) + ((lane >> 4) << 2);
  v8u b;
#pragma unroll
  for (int v = 0; v < 8; ++v) b[v] = Xp[(cpb + v + (v & 4)) * PPAD + p];
  return b;
}

// Projection over the full K=128 reduction for TWO output tiles at once:
// two independent accumulate chains (shared A) -> 2x WMMA-level ILP.
static __device__ __forceinline__ void proj_pair(const v8u A[4],
                                                 const unsigned* __restrict__ Xp,
                                                 int p0, int p1, int lane,
                                                 v8f& acc0, v8f& acc1){
#pragma unroll
  for (int kk = 0; kk < 4; ++kk){
    v8u b0 = loadB_Xp(Xp, 32 * kk, p0, lane);
    v8u b1 = loadB_Xp(Xp, 32 * kk, p1, lane);
    acc0 = wmma_bf16(A[kk], b0, acc0);
    acc1 = wmma_bf16(A[kk], b1, acc1);
  }
}
static __device__ __forceinline__ void proj_one(const v8u A[4],
                                                const unsigned* __restrict__ Xp,
                                                int p, int lane, v8f& acc){
#pragma unroll
  for (int kk = 0; kk < 4; ++kk)
    acc = wmma_bf16(A[kk], loadB_Xp(Xp, 32 * kk, p, lane), acc);
}

// Store one projected K tile with rel bias, packed d-pairs.
// m==0 covers d in [0,16) -> rel_h rows; m==1 -> rel_w cols (uniform select).
static __device__ __forceinline__ void storeK(unsigned* __restrict__ KpH,
                                              const float* __restrict__ sRelH,
                                              const float* __restrict__ sRelW,
                                              int m, int p, int dbase, const v8f& acc){
  int rr = p / WIN;      if (rr > 13) rr = 13;   // clamp for pad pixels
  int cc = p - rr * WIN; if (cc > 13) cc = 13;   // (masked later anyway)
  const float* bbase = m ? (sRelW + cc * 16 - 16) : (sRelH + rr * 16);
#pragma unroll
  for (int t = 0; t < 4; ++t){
    const int d1 = dbase + 2 * t;
    KpH[(d1 >> 1) * PPAD + p] =
        packbf(acc[2 * t] + bbase[d1], acc[2 * t + 1] + bbase[d1 + 1]);
  }
}
static __device__ __forceinline__ void storeV(unsigned short* __restrict__ VtH,
                                              int p, int dbase, const v8f& acc){
#pragma unroll
  for (int t = 0; t < 4; ++t)
    store2_bf16(&VtH[(dbase + 2 * t)     * VSTR + p],
                &VtH[(dbase + 2 * t + 1) * VSTR + p],
                acc[2 * t], acc[2 * t + 1]);
}
static __device__ __forceinline__ void storeQ(unsigned* __restrict__ QbH32w,
                                              int i, int dbase, float scale,
                                              const v8f& acc){
  const int ubase = i * (QSTR / 2) + (dbase >> 1);
#pragma unroll
  for (int t = 0; t < 4; ++t)
    QbH32w[ubase + t] = packbf(acc[2 * t] * scale, acc[2 * t + 1] * scale);
}

__global__ __launch_bounds__(128, 1)
void halo_attn_fused(const float* __restrict__ x, const float* __restrict__ W,
                     const float* __restrict__ relH, const float* __restrict__ relW,
                     float* __restrict__ out)
{
  constexpr int OFF_RELH = 0;
  constexpr int OFF_RELW = 896;
  constexpr int OFF_XP   = 2048;
  constexpr int OFF_KP   = OFF_XP + 64 * PPAD * 4;        //  55808
  constexpr int OFF_VT   = OFF_KP + 4 * 16 * PPAD * 4;    // 109568
  constexpr int OFF_QB   = OFF_VT + 4 * DIM * VSTR * 2;   // 167424
  constexpr int OFF_AT   = OFF_QB + 4 * 64 * QSTR * 2;    // 184832
  // total = OFF_AT + 4*16*ASTR*2 = 213760 bytes  (<= 320KB WGP LDS)

  extern __shared__ char smem[];
  float*          sRelH = (float*)(smem + OFF_RELH);
  float*          sRelW = (float*)(smem + OFF_RELW);
  unsigned*       Xp    = (unsigned*)(smem + OFF_XP);        // [64 cpair][PPAD]
  unsigned*       Kp    = (unsigned*)(smem + OFF_KP);        // [head][16 dpair][PPAD]
  unsigned short* Vt    = (unsigned short*)(smem + OFF_VT);  // [head][32 d][VSTR]
  unsigned short* Qb    = (unsigned short*)(smem + OFF_QB);  // [head][64 i][QSTR]
  unsigned short* At    = (unsigned short*)(smem + OFF_AT);  // [head][16 i][ASTR]

  const int tid  = threadIdx.x;
  const int lane = tid & 31;
  const int head = tid >> 5;
  const int nww = blockIdx.x, nhh = blockIdx.y, bb = blockIdx.z;

  // ---------------- Phase 0: cooperative staging ----------------
  for (int e = tid; e < WIN * 16; e += 128){ sRelH[e] = relH[e]; sRelW[e] = relW[e]; }
  // zero Xp pad columns (p in [196,210))
  for (int e = tid; e < 64 * (PPAD - PIX); e += 128){
    int cp = e / (PPAD - PIX); int p = PIX + e % (PPAD - PIX);
    Xp[cp * PPAD + p] = 0u;
  }
  // zero Vt pad keys (j in [196,224)) so padded-P x V contributes nothing
  for (int e = tid; e < 4 * DIM * (JPAD - PIX); e += 128){
    int j = PIX + e % (JPAD - PIX); int rt = e / (JPAD - PIX);
    Vt[rt * VSTR + j] = 0;
  }
  // zero At pad keys (j in [208,224)) - never written by softmax
  for (int e = tid; e < 4 * 16 * 16; e += 128){
    int j = 208 + (e & 15); int rt = e >> 4;
    At[rt * ASTR + j] = 0;
  }
  // stage x halo window -> packed bf16 channel pairs (zero-padded at borders)
  const int gh0 = nhh * 8 - 3, gw0 = nww * 8 - 3;
  const float* xb = x + (size_t)bb * CH * 128 * 128;
  for (int e = tid; e < 64 * PIX; e += 128){
    int cp = e / PIX, p = e - cp * PIX;
    int r = p / WIN, cc = p - r * WIN;
    int gh = gh0 + r, gw = gw0 + cc;
    float v0 = 0.f, v1 = 0.f;
    if ((unsigned)gh < 128u && (unsigned)gw < 128u){
      const float* px = xb + ((size_t)(2 * cp) * 128 + gh) * 128 + gw;
      v0 = px[0];
      v1 = px[128 * 128];
    }
    Xp[cp * PPAD + p] = packbf(v0, v1);
  }
  __syncthreads();

  unsigned*       KpH = Kp + head * 16 * PPAD;
  unsigned short* VtH = Vt + head * DIM * VSTR;
  unsigned short* QbH = Qb + head * 64 * QSTR;
  unsigned short* AtH = At + head * 16 * ASTR;

  // prefetch this head's q/k/v weight rows into cache (global_prefetch_b8)
#pragma unroll
  for (int blk = 0; blk < 3; ++blk){
    const float* wp = W + (blk * CH + head * DIM) * CH;
    for (int q = lane; q < (DIM * CH) / 32; q += 32)
      __builtin_prefetch(wp + q * 32, 0, 1);
  }

  const int pcol = lane & 15;

  // ---------------- Phase A: per-head projections ----------------
  // K projection: rows [128 + head*32)
  {
    const int row0 = CH + head * DIM;
#pragma unroll 1
    for (int m = 0; m < 2; ++m){
      const int d0 = 16 * m;
      const int dbase = d0 + ((lane >> 4) << 3);
      v8u A[4];
#pragma unroll
      for (int kk = 0; kk < 4; ++kk) A[kk] = loadA_W(W, row0 + d0, 32 * kk, lane);
#pragma unroll 1
      for (int ntp = 0; ntp < 6; ++ntp){
        const int p0 = 32 * ntp + pcol, p1 = p0 + 16;
        v8f acc0 = V8F_ZERO, acc1 = V8F_ZERO;
        proj_pair(A, Xp, p0, p1, lane, acc0, acc1);
        storeK(KpH, sRelH, sRelW, m, p0, dbase, acc0);
        storeK(KpH, sRelH, sRelW, m, p1, dbase, acc1);
      }
      { // tail tile nt = 12
        const int p = 192 + pcol;
        v8f acc = V8F_ZERO;
        proj_one(A, Xp, p, lane, acc);
        storeK(KpH, sRelH, sRelW, m, p, dbase, acc);
      }
    }
  }
  // V projection: rows [256 + head*32), stored as V^T bf16 [d][j].
  {
    const int row0 = 2 * CH + head * DIM;
#pragma unroll 1
    for (int m = 0; m < 2; ++m){
      const int d0 = 16 * m;
      const int dbase = d0 + ((lane >> 4) << 3);
      v8u A[4];
#pragma unroll
      for (int kk = 0; kk < 4; ++kk) A[kk] = loadA_W(W, row0 + d0, 32 * kk, lane);
#pragma unroll 1
      for (int ntp = 0; ntp < 6; ++ntp){
        const int p0 = 32 * ntp + pcol, p1 = p0 + 16;
        v8f acc0 = V8F_ZERO, acc1 = V8F_ZERO;
        proj_pair(A, Xp, p0, p1, lane, acc0, acc1);
        storeV(VtH, p0, dbase, acc0);
        storeV(VtH, p1, dbase, acc1);
      }
      { // tail tile nt = 12
        const int p = 192 + pcol;
        v8f acc = V8F_ZERO;
        proj_one(A, Xp, p, lane, acc);
        storeV(VtH, p, dbase, acc);
      }
    }
  }
  // Q projection: rows [head*32), center 8x8 block pixels.
  // Scale folds d^-0.5 (reference) AND log2(e) (base-2 softmax) into Q.
  {
    const float QSCALE = 0.17677669529663689f * 1.4426950408889634f;
    unsigned* QbH32w = (unsigned*)QbH;
    const int row0 = head * DIM;
#pragma unroll 1
    for (int m = 0; m < 2; ++m){
      const int d0 = 16 * m;
      const int dbase = d0 + ((lane >> 4) << 3);
      v8u A[4];
#pragma unroll
      for (int kk = 0; kk < 4; ++kk) A[kk] = loadA_W(W, row0 + d0, 32 * kk, lane);
#pragma unroll 1
      for (int ntp = 0; ntp < 2; ++ntp){
        const int i0q = 32 * ntp + pcol, i1q = i0q + 16;
        const int pp0 = (3 + (i0q >> 3)) * WIN + 3 + (i0q & 7);
        const int pp1 = (3 + (i1q >> 3)) * WIN + 3 + (i1q & 7);
        v8f acc0 = V8F_ZERO, acc1 = V8F_ZERO;
        proj_pair(A, Xp, pp0, pp1, lane, acc0, acc1);
        storeQ(QbH32w, i0q, dbase, QSCALE, acc0);
        storeQ(QbH32w, i1q, dbase, QSCALE, acc1);
      }
    }
  }

  // ---------------- Phase B: attention (per 16-query block) ----------------
  const unsigned* QbH32 = (const unsigned*)QbH;   // u32 row stride 17
  const unsigned* VtH32 = (const unsigned*)VtH;   // u32 row stride 113
  const unsigned* AtH32 = (const unsigned*)AtH;   // u32 row stride 113

#pragma unroll 1
  for (int ib = 0; ib < 4; ++ib){
    const int i0 = 16 * ib;
    // A = Q tile (one tile covers all 32 dims, so each sim tile is ONE wmma)
    v8u aq;
    {
      const int base = (i0 + pcol) * 17 + ((lane >> 4) << 2);
#pragma unroll
      for (int t = 0; t < 4; ++t){ aq[t] = QbH32[base + t]; aq[4 + t] = QbH32[base + 8 + t]; }
    }
    // sim = Q @ K^T over 13 key tiles (196 -> 208 padded); logits already *log2e
    v8f s[13];
    const int kpb = ((lane >> 4) << 2);
#pragma unroll
    for (int nt = 0; nt < 13; ++nt){
      v8u bk;
#pragma unroll
      for (int v = 0; v < 8; ++v) bk[v] = KpH[(kpb + v + (v & 4)) * PPAD + 16 * nt + pcol];
      s[nt] = wmma_bf16(aq, bk, V8F_ZERO);
    }
    // mask padded keys j in [196,208)
    if (192 + pcol >= PIX){
#pragma unroll
      for (int r = 0; r < 8; ++r) s[12][r] = -1e30f;
    }
    // base-2 softmax over j: per-row max/sum, rows live in 16-lane halves
    float mx[8], sm[8];
#pragma unroll
    for (int r = 0; r < 8; ++r){
      float mv = s[0][r];
#pragma unroll
      for (int nt = 1; nt < 13; ++nt) mv = fmaxf(mv, s[nt][r]);
#pragma unroll
      for (int off = 1; off < 16; off <<= 1) mv = fmaxf(mv, __shfl_xor(mv, off, 32));
      mx[r] = mv;
    }
#pragma unroll
    for (int r = 0; r < 8; ++r){
      float acc = 0.f;
#pragma unroll
      for (int nt = 0; nt < 13; ++nt){
        float e = EXP2F(s[nt][r] - mx[r]);   // v_exp_f32 directly (base-2)
        s[nt][r] = e;
        acc += e;
      }
#pragma unroll
      for (int off = 1; off < 16; off <<= 1) acc += __shfl_xor(acc, off, 32);
      sm[r] = 1.0f / acc;
    }
    // store P in bf16 (pairs over j are contiguous -> B tiles are 8 ds_load_b32)
    const int irow = ((lane >> 4) << 3);
#pragma unroll
    for (int nt = 0; nt < 13; ++nt){
      const int j = 16 * nt + pcol;
#pragma unroll
      for (int t = 0; t < 4; ++t)
        store2_bf16(&AtH[(irow + 2 * t)     * ASTR + j],
                    &AtH[(irow + 2 * t + 1) * ASTR + j],
                    s[nt][2 * t] * sm[2 * t], s[nt][2 * t + 1] * sm[2 * t + 1]);
    }
    // out^T = V^T @ P^T  (K = 224 padded keys, 7 wmma steps, 2 d-tiles)
    v8f o[2];
    o[0] = V8F_ZERO;
    o[1] = V8F_ZERO;
#pragma unroll 1
    for (int ks = 0; ks < 7; ++ks){
      const int jp = 16 * ks;                    // j0/2
      v8u bt;
      {
        const int pb = jp + ((lane >> 4) << 2);
#pragma unroll
        for (int v = 0; v < 8; ++v) bt[v] = AtH32[pcol * 113 + pb + v + (v & 4)];
      }
#pragma unroll
      for (int dm = 0; dm < 2; ++dm){
        v8u av;
        const int d  = 16 * dm + pcol;
        const int pb = jp + ((lane >> 4) << 2);
#pragma unroll
        for (int t = 0; t < 4; ++t){
          av[t]     = VtH32[d * 113 + pb + t];
          av[4 + t] = VtH32[d * 113 + pb + 8 + t];
        }
        o[dm] = wmma_bf16(av, bt, o[dm]);
      }
    }
    // write out[b, head*32+d, gh, gw] straight from D-tile layout
    const int i  = i0 + pcol;
    const int gh = nhh * 8 + (i >> 3);
    const int gw = nww * 8 + (i & 7);
    float* ob = out + (size_t)bb * CH * 128 * 128 + (size_t)gh * 128 + gw;
#pragma unroll
    for (int dm = 0; dm < 2; ++dm){
      const int cb = head * DIM + 16 * dm + irow;
#pragma unroll
      for (int r = 0; r < 8; ++r)
        ob[(size_t)(cb + r) * 128 * 128] = o[dm][r];
    }
  }
}

extern "C" void kernel_launch(void* const* d_in, const int* in_sizes, int n_in,
                              void* d_out, int out_size, void* d_ws, size_t ws_size,
                              hipStream_t stream)
{
  (void)in_sizes; (void)n_in; (void)out_size; (void)d_ws; (void)ws_size;
  const float* x    = (const float*)d_in[0];
  const float* qkvw = (const float*)d_in[1];
  const float* relh = (const float*)d_in[2];
  const float* relw = (const float*)d_in[3];
  float*       out  = (float*)d_out;

  const int smem_bytes = 213760;  // 209 KB of the 320 KB WGP LDS
  (void)hipFuncSetAttribute(reinterpret_cast<const void*>(halo_attn_fused),
                            hipFuncAttributeMaxDynamicSharedMemorySize, smem_bytes);
  // grid: (nw=16, nh=16, b=8) windows; 128 threads = 4 wave32 (one head each)
  halo_attn_fused<<<dim3(16, 16, 8), 128, smem_bytes, stream>>>(
      x, qkvw, relh, relw, out);
}
